// COLA_TTVSRNet_34935263985696
// MI455X (gfx1250) — matrix operand order repaired and checked by reading, not compiled, not run on hardware
//
#include <hip/hip_runtime.h>
#include <hip/hip_fp16.h>
#include <cstdint>

#define HS_  48
#define FN   2304          // 48*48
#define FL   1024          // C*S*S
#define TT   8
#define GG   4
#define CGC  256
#define HH   192
#define WW   192
#define HW   (HH*WW)
#define KP   16            // padded taps per input channel (9 real + 7 zero)
#define KTOT (192*KP)      // 3072
#define PW   194           // padded width for fusion conv input
#define PHW  (PW*PW)

typedef __attribute__((ext_vector_type(16))) _Float16     v16h;
typedef __attribute__((ext_vector_type(8)))  float        v8f;
typedef __attribute__((ext_vector_type(4)))  unsigned int u32x4;
typedef __attribute__((ext_vector_type(8)))  int          i32x8;
typedef __attribute__((ext_vector_type(4)))  int          i32x4;

__device__ __forceinline__ int iclamp(int v, int lo, int hi) { return v < lo ? lo : (v > hi ? hi : v); }

__device__ __forceinline__ float block_reduce_sum(float v, float* red) {
  int tid = threadIdx.x;
  red[tid] = v; __syncthreads();
  for (int s = 128; s > 0; s >>= 1) { if (tid < s) red[tid] += red[tid + s]; __syncthreads(); }
  float r = red[0]; __syncthreads();
  return r;
}

// K0: nearest-gather table per (t, f): gather offset, validity, valid/max(||oi||,eps)
__global__ void k_prep(const float* __restrict__ loc, const float* __restrict__ idx1,
                       int* __restrict__ goff, float* __restrict__ validf, float* __restrict__ cfn) {
  __shared__ float red[256];
  int t = blockIdx.x, f = blockIdx.y;
  float lx = loc[(t*2+0)*FN + f];
  float ly = loc[(t*2+1)*FN + f];
  int ixn = (int)rintf(lx), iyn = (int)rintf(ly);
  bool valid = (ixn >= 0 && ixn < HS_ && iyn >= 0 && iyn < HS_);
  int p = iclamp(iyn,0,HS_-1)*HS_ + iclamp(ixn,0,HS_-1);
  const float* base = idx1 + (size_t)t*FL*FN + p;
  float ss = 0.f;
  for (int k = 0; k < 4; ++k) {
    float v = base[(size_t)(threadIdx.x + k*256)*FN];
    ss += v*v;
  }
  float tot = block_reduce_sum(ss, red);
  if (threadIdx.x == 0) {
    goff[t*FN+f]   = p;
    validf[t*FN+f] = valid ? 1.f : 0.f;
    cfn[t*FN+f]    = valid ? (1.f / fmaxf(sqrtf(tot), 1e-12f)) : 0.f;
  }
}

// K1: cn[c'][f] = unfold(curr) L2-normalized over c' per f
__global__ void k_cn(const float* __restrict__ curr, float* __restrict__ cn) {
  __shared__ float red[256];
  int f = blockIdx.x;
  int y = f / HS_, x = f % HS_;
  float vals[4]; float ss = 0.f;
  for (int k = 0; k < 4; ++k) {
    int cp = threadIdx.x + k*256;
    int cch = cp >> 4, sy = (cp >> 2) & 3, sx = cp & 3;
    float v = curr[(size_t)cch*HW + (size_t)(y*4+sy)*WW + (x*4+sx)];
    vals[k] = v; ss += v*v;
  }
  float tot = block_reduce_sum(ss, red);
  float inv = 1.f / fmaxf(sqrtf(tot), 1e-12f);
  for (int k = 0; k < 4; ++k) {
    int cp = threadIdx.x + k*256;
    cn[(size_t)cp*FN + f] = vals[k]*inv;
  }
}

// K2: offset head (grouped 5x5 dw conv + LN + GELU + 1x1) -> sampling grid -> composed tap table
__global__ void k_head(const float* __restrict__ cn, const float* __restrict__ idx1,
                       const int* __restrict__ goff, const float* __restrict__ cfn,
                       const float* __restrict__ validf,
                       const float* __restrict__ wtdw, const float* __restrict__ btdw,
                       const float* __restrict__ lng, const float* __restrict__ lnb,
                       const float* __restrict__ wtpw,
                       int* __restrict__ gi, float* __restrict__ wmat, float* __restrict__ wsg) {
  __shared__ float red[256];
  __shared__ float sh2[2];
  int b = blockIdx.x, f = blockIdx.y;
  int t = b >> 2, g = b & 3;
  int y = f / HS_, x = f % HS_;
  int cg = threadIdx.x;
  const int*   go_t = goff + t*FN;
  const float* cf_t = cfn  + t*FN;
  // grouped conv: group cg consumes concat-channels 2cg, 2cg+1 of [qo(256) | ko(256)]
  int i0 = 2*cg, i1 = 2*cg+1;
  float acc = btdw[cg];
  const float* wrow = wtdw + cg*50;
  for (int dy = 0; dy < 5; ++dy) {
    int yy = y + dy - 2;
    if ((unsigned)yy >= (unsigned)HS_) continue;
    for (int dx = 0; dx < 5; ++dx) {
      int xx = x + dx - 2;
      if ((unsigned)xx >= (unsigned)HS_) continue;
      int fp = yy*HS_ + xx;
      float v0 = (i0 < 256) ? cn[(size_t)(g*CGC + i0)*FN + fp]
                            : idx1[((size_t)t*FL + g*CGC + (i0-256))*FN + go_t[fp]] * cf_t[fp];
      float v1 = (i1 < 256) ? cn[(size_t)(g*CGC + i1)*FN + fp]
                            : idx1[((size_t)t*FL + g*CGC + (i1-256))*FN + go_t[fp]] * cf_t[fp];
      acc += wrow[dy*5+dx]*v0 + wrow[25 + dy*5+dx]*v1;
    }
  }
  float s1r = block_reduce_sum(acc, red);
  float s2r = block_reduce_sum(acc*acc, red);
  float mean = s1r * (1.f/256.f);
  float var  = s2r * (1.f/256.f) - mean*mean;
  float xn = (acc - mean) * rsqrtf(var + 1e-5f) * lng[cg] + lnb[cg];
  float xg = 0.5f * xn * (1.f + erff(xn * 0.70710678118f));  // exact GELU
  float pw0 = block_reduce_sum(xg * wtpw[cg],       red);
  float pw1 = block_reduce_sum(xg * wtpw[256 + cg], red);
  if (threadIdx.x == 0) {
    float posy = tanhf(pw0) * (2.0f / 48.f) + ((0.5f + y) / 48.f * 2.f - 1.f);
    float posx = tanhf(pw1) * (2.0f / 48.f) + ((0.5f + x) / 48.f * 2.f - 1.f);
    sh2[0] = posx; sh2[1] = posy;
  }
  __syncthreads();
  if (threadIdx.x < 4) {
    float ix = (sh2[0] + 1.f) * 0.5f * (HS_-1);
    float iy = (sh2[1] + 1.f) * 0.5f * (HS_-1);
    float x0 = floorf(ix), y0 = floorf(iy);
    float wx = ix - x0,    wy = iy - y0;
    int j = threadIdx.x;
    int dy = j >> 1, dx = j & 1;
    int xi = (int)x0 + dx, yi = (int)y0 + dy;
    float wgt = (dy ? wy : 1.f-wy) * (dx ? wx : 1.f-wx);
    bool ok = (xi >= 0 && xi < HS_ && yi >= 0 && yi < HS_);
    int p = iclamp(yi,0,HS_-1)*HS_ + iclamp(xi,0,HS_-1);
    float wb = ok ? wgt : 0.f;
    size_t o = ((size_t)b*FN + f)*4 + j;
    gi[o]   = go_t[p];                       // compose bilinear with nearest-gather
    wmat[o] = wb * cf_t[p];                  // includes validity * inv-norm (key path)
    wsg[o]  = wb * validf[t*FN + p];         // validity only (value path)
  }
}

// K3: mat[f,t] = <sampled key, cn> over 1024 ch; max/argmax over t
__global__ void k_mat(const float* __restrict__ cn, const float* __restrict__ idx1,
                      const int* __restrict__ gi, const float* __restrict__ wmat,
                      float* __restrict__ csoft, int* __restrict__ cidx) {
  __shared__ float red[256];
  int f = blockIdx.x;
  int tid = threadIdx.x;
  float best = 0.f; int bi = 0;
  for (int t = 0; t < TT; ++t) {
    float s = 0.f;
    for (int g = 0; g < GG; ++g) {
      int b = t*GG + g;
      int c = g*CGC + tid;
      const float* it = idx1 + ((size_t)t*FL + c)*FN;
      size_t o = ((size_t)b*FN + f)*4;
      float v = wmat[o+0]*it[gi[o+0]] + wmat[o+1]*it[gi[o+1]]
              + wmat[o+2]*it[gi[o+2]] + wmat[o+3]*it[gi[o+3]];
      s += v * cn[(size_t)c*FN + f];
    }
    float tot = block_reduce_sum(s, red);
    if (tid == 0 && (t == 0 || tot > best)) { best = tot; bi = t; }
  }
  if (tid == 0) { csoft[f] = best; cidx[f] = bi; }
}

// K4: sample s1/s2/s3 through composed taps at argmax frame, fold into padded f16 conv input
__global__ void k_sg(const float* __restrict__ s1, const float* __restrict__ s2, const float* __restrict__ s3,
                     const int* __restrict__ cidx, const int* __restrict__ gi, const float* __restrict__ wsg,
                     __half* __restrict__ inP) {
  size_t id = (size_t)blockIdx.x * blockDim.x + threadIdx.x;
  if (id >= (size_t)3*FL*FN) return;
  int f   = (int)(id % FN);
  int cp  = (int)((id / FN) % FL);
  int set = (int)(id / ((size_t)FL*FN));
  int t = cidx[f];
  int g = cp >> 8;
  size_t o = ((size_t)(t*GG + g)*FN + f)*4;
  const float* src = (set==0) ? s1 : (set==1 ? s2 : s3);
  const float* sc = src + ((size_t)t*FL + cp)*FN;
  float v = wsg[o+0]*sc[gi[o+0]] + wsg[o+1]*sc[gi[o+1]]
          + wsg[o+2]*sc[gi[o+2]] + wsg[o+3]*sc[gi[o+3]];
  int y = f / HS_, x = f % HS_;
  int cch = cp >> 4, sy = (cp >> 2) & 3, sx = cp & 3;
  int hh = y*4 + sy, ww = x*4 + sx;
  inP[(size_t)(set*64 + cch)*PHW + (size_t)(hh+1)*PW + (ww+1)] = __float2half(v);
}

// K5: fusion weights -> f16, K-major [k][m], taps padded 9 -> 16 with zeros
__global__ void k_wprep(const float* __restrict__ wfus, __half* __restrict__ wA) {
  int id = blockIdx.x * blockDim.x + threadIdx.x;
  if (id >= KTOT*64) return;
  int m = id % 64;
  int k = id / 64;
  int cin = k >> 4, tap = k & 15;
  float v = (tap < 9) ? wfus[((size_t)m*192 + cin)*9 + tap] : 0.f;
  wA[(size_t)k*64 + m] = __float2half(v);
}

// K6: 3x3 fusion conv as implicit-im2col WMMA GEMM (M=64, K=3072, N=64 px/block)
//     + fused bias, csoft_f scaling and anchor residual. TDM stages 4KB weight slices.
__global__ void __launch_bounds__(256) k_fusion(const __half* __restrict__ inPh, const __half* __restrict__ wAh,
                                                const float* __restrict__ bfus, const float* __restrict__ csoft,
                                                const float* __restrict__ anchor, float* __restrict__ out) {
  __shared__ _Float16 ldsA[32*64];        // K-slice of weights, [k][m]
  __shared__ _Float16 ldsB[2][3][72];     // input halo: 2 ch x 3 rows x 66(+pad) cols
  const _Float16* inP = (const _Float16*)inPh;
  int tid  = threadIdx.x;
  int lane = tid & 31;
  int wv   = tid >> 5;
  int mr   = wv >> 1;           // M-tile row (0..3) -> out channels mr*16..+15
  int ncb  = (wv & 1) * 2;      // first of two N-tiles for this wave
  int y    = blockIdx.y;
  int x0g  = blockIdx.x * 64;

  v8f acc0 = {}; v8f acc1 = {};
  int mloc = mr*16 + (lane & 15);
  int kset = lane >> 4;

  uint32_t ldsAoff = (uint32_t)(uintptr_t)(&ldsA[0]);   // low 32 bits of generic addr = LDS offset
  uint64_t wbase   = (uint64_t)(uintptr_t)wAh;

  for (int it = 0; it < 96; ++it) {
    __syncthreads();
    if (tid < 32) {
      // TDM: contiguous 32x64 f16 slice (4KB) of K-major weights -> LDS
      uint64_t ga = wbase + (uint64_t)it * 4096u;
      u32x4 g0;
      g0[0] = 1u;                                            // count=1, normal mode
      g0[1] = ldsAoff;                                       // lds_addr
      g0[2] = (unsigned)(ga & 0xffffffffu);                  // global_addr[31:0]
      g0[3] = (unsigned)((ga >> 32) & 0x01ffffffu) | (2u<<30); // global_addr[56:32] | type=2
      i32x8 g1;
      g1[0] = (1 << 16);            // data_size = 2 bytes
      g1[1] = (int)(2048u << 16);   // tensor_dim0 low16 (=2048)
      g1[2] = (1 << 16);            // tensor_dim1 low16 (=1)
      g1[3] = (int)(2048u << 16);   // tile_dim0 = 2048
      g1[4] = 1;                    // tile_dim1 = 1
      g1[5] = 2048;                 // tensor_dim0_stride
      g1[6] = 0; g1[7] = 0;
      i32x4 z4 = {0,0,0,0};
#if defined(__clang_major__) && (__clang_major__ >= 23)
      i32x8 z8 = {0,0,0,0,0,0,0,0};
      __builtin_amdgcn_tensor_load_to_lds(g0, g1, z4, z4, z8, 0);
#else
      __builtin_amdgcn_tensor_load_to_lds(g0, g1, z4, z4, 0);
#endif
      __builtin_amdgcn_s_wait_tensorcnt(0);
    }
    // B halo: channels 2*it, 2*it+1; padded rows y..y+2; padded cols x0g..x0g+65
    int c0 = it * 2;
    for (int i = tid; i < 396; i += 256) {
      int cc = i / 198;
      int r  = (i % 198) / 66;
      int xx = i % 66;
      ldsB[cc][r][xx] = inP[(size_t)(c0 + cc)*PHW + (size_t)(y + r)*PW + (x0g + xx)];
    }
    __syncthreads();

    // A fragment: lane L holds M = L%16; K groups split by L/16 per ISA 16-bit A layout
    v16h a;
    #pragma unroll
    for (int i2 = 0; i2 < 8; ++i2) {
      int klo = (i2 < 4) ? (kset*8 + 2*i2) : (16 + kset*8 + 2*(i2-4));
      a[2*i2]   = ldsA[klo*64 + mloc];
      a[2*i2+1] = ldsA[(klo+1)*64 + mloc];
    }
    #pragma unroll
    for (int j = 0; j < 2; ++j) {
      int nc = ncb + j;
      int xloc = nc*16 + (lane & 15);
      v16h bf;
      #pragma unroll
      for (int i2 = 0; i2 < 16; ++i2) {
        int kl = (i2 < 8) ? (kset*8 + i2) : (16 + kset*8 + (i2-8));          // K within 32-slice
        int cc = kl >> 4;               // which of the 2 channels
        int tap = kl & 15;              // padded tap id
        _Float16 v = (_Float16)0.f;
        if (tap < 9) v = ldsB[cc][tap/3][xloc + (tap%3)];
        bf[i2] = v;
      }
      if (j == 0)
        acc0 = __builtin_amdgcn_wmma_f32_16x16x32_f16(false, a, false, bf, (short)0, acc0, false, false);
      else
        acc1 = __builtin_amdgcn_wmma_f32_16x16x32_f16(false, a, false, bf, (short)0, acc1, false, false);
    }
  }
  // Epilogue per documented C/D layout: vgpr r -> M=r (lanes 0-15) / M=r+8 (lanes 16-31)
  int nloc = lane & 15;
  int moff = (lane >= 16) ? 8 : 0;
  #pragma unroll
  for (int j = 0; j < 2; ++j) {
    int xg = x0g + (ncb + j)*16 + nloc;
    float cs = csoft[(y >> 2)*HS_ + (xg >> 2)];
    v8f acc = j ? acc1 : acc0;
    #pragma unroll
    for (int r = 0; r < 8; ++r) {
      int m = mr*16 + r + moff;
      size_t oidx = (size_t)m*HW + (size_t)y*WW + xg;
      out[oidx] = (acc[r] + bfus[m]) * cs + anchor[oidx];
    }
  }
}

extern "C" void kernel_launch(void* const* d_in, const int* in_sizes, int n_in,
                              void* d_out, int out_size, void* d_ws, size_t ws_size,
                              hipStream_t stream) {
  (void)in_sizes; (void)n_in; (void)out_size; (void)ws_size;
  const float* curr   = (const float*)d_in[0];
  const float* idx1   = (const float*)d_in[1];
  const float* anchor = (const float*)d_in[2];
  const float* s1     = (const float*)d_in[3];
  const float* s2     = (const float*)d_in[4];
  const float* s3     = (const float*)d_in[5];
  const float* loc    = (const float*)d_in[6];
  const float* wtdw   = (const float*)d_in[7];
  const float* btdw   = (const float*)d_in[8];
  const float* lng    = (const float*)d_in[9];
  const float* lnb    = (const float*)d_in[10];
  const float* wtpw   = (const float*)d_in[11];
  const float* wfus   = (const float*)d_in[12];
  const float* bfus   = (const float*)d_in[13];
  float* out = (float*)d_out;

  char* p = (char*)d_ws;
  auto carve = [&](size_t bytes) { char* r = p; p += (bytes + 255) & ~(size_t)255; return r; };
  int*    goff   = (int*)   carve((size_t)TT*FN*sizeof(int));
  float*  validf = (float*) carve((size_t)TT*FN*sizeof(float));
  float*  cfn    = (float*) carve((size_t)TT*FN*sizeof(float));
  float*  cn     = (float*) carve((size_t)FL*FN*sizeof(float));
  int*    gi     = (int*)   carve((size_t)32*FN*4*sizeof(int));
  float*  wmat   = (float*) carve((size_t)32*FN*4*sizeof(float));
  float*  wsg    = (float*) carve((size_t)32*FN*4*sizeof(float));
  float*  csoft  = (float*) carve((size_t)FN*sizeof(float));
  int*    cidxb  = (int*)   carve((size_t)FN*sizeof(int));
  __half* inP    = (__half*)carve((size_t)192*PHW*sizeof(__half));
  __half* wA     = (__half*)carve((size_t)KTOT*64*sizeof(__half));

  hipMemsetAsync(inP, 0, (size_t)192*PHW*sizeof(__half), stream);

  k_prep <<<dim3(TT, FN), 256, 0, stream>>>(loc, idx1, goff, validf, cfn);
  k_cn   <<<FN, 256, 0, stream>>>(curr, cn);
  k_head <<<dim3(32, FN), 256, 0, stream>>>(cn, idx1, goff, cfn, validf,
                                            wtdw, btdw, lng, lnb, wtpw, gi, wmat, wsg);
  k_mat  <<<FN, 256, 0, stream>>>(cn, idx1, gi, wmat, csoft, cidxb);
  size_t nsg = (size_t)3*FL*FN;
  k_sg   <<<(unsigned)((nsg + 255)/256), 256, 0, stream>>>(s1, s2, s3, cidxb, gi, wsg, inP);
  k_wprep<<<(KTOT*64 + 255)/256, 256, 0, stream>>>(wfus, wA);
  k_fusion<<<dim3(3, HH), 256, 0, stream>>>(inP, wA, bfus, csoft, anchor, out);
}
